// InfiniteDurationRegulator_84722524881488
// MI455X (gfx1250) — matrix operand order repaired and checked by reading, not compiled, not run on hardware
//
#include <hip/hip_runtime.h>
#include <hip/hip_bf16.h>
#include <cmath>

// ---------------------------------------------------------------- problem dims
#define B_    8
#define T_    2048
#define TD_   512
#define SD_   128
#define HD_   512
#define SS_   64
#define ID_   1024          // 2*HD
#define M_    (B_ * T_)     // 16384 token rows
#define MAXDUR_ 10
#define RLEN_ (T_ * MAXDUR_) // 20480

typedef __bf16 bf16_t;
typedef __attribute__((ext_vector_type(8)))  __bf16 v8bf;
typedef __attribute__((ext_vector_type(16))) __bf16 v16bf;
typedef __attribute__((ext_vector_type(8)))  float  v8f;
typedef __attribute__((ext_vector_type(8)))  int    v8i;

__device__ __forceinline__ float siluf(float x) { return x / (1.0f + __expf(-x)); }

// ---------------------------------------------------------------- WMMA helpers
// A fragment (16x32 bf16), ISA 7.12.2: lane l -> row M=l&15;
// elems 0..7 = K (8*(l>>4))+0..7 ; elems 8..15 = same +16.
__device__ __forceinline__ v16bf load_a_frag(const bf16_t* __restrict__ aptr_k0) {
    v8bf lo = *(const v8bf*)(aptr_k0);
    v8bf hi = *(const v8bf*)(aptr_k0 + 16);
    v16bf r;
#pragma unroll
    for (int i = 0; i < 8; ++i) { r[i] = lo[i]; r[i + 8] = hi[i]; }
    return r;
}
// B fragment (32x16 bf16) from pre-transposed WT[N][K]:
// lane l -> col N=l&15; elems 0..7 = K (8*(l>>4))+0..7 ; elems 8..15 = same +16.
__device__ __forceinline__ v16bf load_b_frag(const bf16_t* __restrict__ bptr) {
    v8bf lo = *(const v8bf*)(bptr);
    v8bf hi = *(const v8bf*)(bptr + 16);
    v16bf r;
#pragma unroll
    for (int i = 0; i < 8; ++i) { r[i] = lo[i]; r[i + 8] = hi[i]; }
    return r;
}
// branchless per-lane zeroing (v_cndmask on 32-bit lanes, no EXEC divergence)
__device__ __forceinline__ v16bf mask_frag(v16bf a, bool valid) {
    v8i ai = __builtin_bit_cast(v8i, a);
#pragma unroll
    for (int i = 0; i < 8; ++i) ai[i] = valid ? ai[i] : 0;
    return __builtin_bit_cast(v16bf, ai);
}
__device__ __forceinline__ v8f wmma_bf16(v16bf a, v16bf b, v8f c) {
    return __builtin_amdgcn_wmma_f32_16x16x32_bf16(false, a, false, b, (short)0, c,
                                                   false, false);
}

// ---------------------------------------------------------------- FiLM MLP
__global__ __launch_bounds__(512) void film_kernel(
    const float* __restrict__ style, const float* __restrict__ w1,
    const float* __restrict__ w2, float* __restrict__ filmp) {
    __shared__ float hid[HD_];
    const int b = blockIdx.x, j = threadIdx.x;
    float s = 0.f;
    for (int k = 0; k < SD_; ++k) s += style[b * SD_ + k] * w1[k * HD_ + j];
    hid[j] = fmaxf(s, 0.f);
    __syncthreads();
    float f = 0.f;
    for (int k = 0; k < HD_; ++k) f += hid[k] * w2[k * TD_ + j];
    filmp[b * TD_ + j] = 1.0f + f;   // store (1 + film) directly
}

// text * (1+film) -> bf16 GEMM operand (8 elems / thread)
__global__ void modulate_kernel(const float* __restrict__ text,
                                const float* __restrict__ filmp,
                                bf16_t* __restrict__ out) {
    size_t i = ((size_t)blockIdx.x * blockDim.x + threadIdx.x) * 8;
    int c = (int)(i & (TD_ - 1));
    int b = (int)(i >> 20);                                    // T_*TD_ = 2^20
    float4 t0 = *(const float4*)(text + i);
    float4 t1 = *(const float4*)(text + i + 4);
    float4 f0 = *(const float4*)(filmp + b * TD_ + c);
    float4 f1 = *(const float4*)(filmp + b * TD_ + c + 4);
    v8bf o;
    o[0] = (bf16_t)(t0.x * f0.x); o[1] = (bf16_t)(t0.y * f0.y);
    o[2] = (bf16_t)(t0.z * f0.z); o[3] = (bf16_t)(t0.w * f0.w);
    o[4] = (bf16_t)(t1.x * f1.x); o[5] = (bf16_t)(t1.y * f1.y);
    o[6] = (bf16_t)(t1.z * f1.z); o[7] = (bf16_t)(t1.w * f1.w);
    *(v8bf*)(out + i) = o;
}

// W[K,N] fp32 (strided) -> WT[N,K] bf16
__global__ void transpose_w_kernel(const float* __restrict__ in,
                                   bf16_t* __restrict__ out, int K, int N,
                                   int rs, int cs, int base) {
    int idx = blockIdx.x * blockDim.x + threadIdx.x;
    if (idx >= N * K) return;
    int n = idx / K, k = idx - n * K;
    out[idx] = (bf16_t)in[base + k * rs + n * cs];
}

// ---------------------------------------------------------------- generic WMMA GEMM
// C[M,N] = A[M,K] (bf16, row-major, lda) x WT[N,K] (bf16). 128x64 block tile,
// 8 waves, each wave 16 rows x 64 cols. K processed in steps of 64 with manual
// double buffering so loads of group k+32 are in flight while group k is in the
// matrix pipe (K is always a multiple of 64 here).
// Epilogues:
//  0 f32     1 bf16     2 relu->bf16   3 relu->f32   4 *alpha->f32
//  5 +res->f32          6 f32 AND bf16
//  7 mamba: ((acc + vecD[col]*aux1) * aux2) -> bf16   (aux2 = silu(gate) already)
template <int EPI>
__global__ __launch_bounds__(256) void gemm_bf16_wmma(
    const bf16_t* __restrict__ A, int lda, const bf16_t* __restrict__ WT, int K,
    float* __restrict__ outf, bf16_t* __restrict__ outb, int ldc, int col0,
    const float* __restrict__ res, const bf16_t* __restrict__ aux1,
    const bf16_t* __restrict__ aux2, const float* __restrict__ vecD, float alpha) {
    const int lane = threadIdx.x & 31;
    const int wave = threadIdx.x >> 5;
    const int m0 = blockIdx.y * 128 + wave * 16;
    const int n0 = blockIdx.x * 64;

    const bf16_t* aptr = A + (size_t)(m0 + (lane & 15)) * lda + ((lane >> 4) << 3);
    const int bcol = lane & 15;
    const int bk = (lane >> 4) << 3;
    const bf16_t* bp0 = WT + (size_t)(n0 + 0 * 16 + bcol) * K + bk;
    const bf16_t* bp1 = WT + (size_t)(n0 + 1 * 16 + bcol) * K + bk;
    const bf16_t* bp2 = WT + (size_t)(n0 + 2 * 16 + bcol) * K + bk;
    const bf16_t* bp3 = WT + (size_t)(n0 + 3 * 16 + bcol) * K + bk;

    v8f acc0 = {}, acc1 = {}, acc2 = {}, acc3 = {};
    // preload group 0
    v16bf aA = load_a_frag(aptr);
    v16bf bA0 = load_b_frag(bp0), bA1 = load_b_frag(bp1),
          bA2 = load_b_frag(bp2), bA3 = load_b_frag(bp3);
    for (int k0 = 0; k0 < K; k0 += 64) {
        // issue loads for group k0+32 (always valid: k0 <= K-64)
        v16bf aB = load_a_frag(aptr + k0 + 32);
        v16bf bB0 = load_b_frag(bp0 + k0 + 32), bB1 = load_b_frag(bp1 + k0 + 32),
              bB2 = load_b_frag(bp2 + k0 + 32), bB3 = load_b_frag(bp3 + k0 + 32);
        acc0 = wmma_bf16(aA, bA0, acc0); acc1 = wmma_bf16(aA, bA1, acc1);
        acc2 = wmma_bf16(aA, bA2, acc2); acc3 = wmma_bf16(aA, bA3, acc3);
        if (k0 + 64 < K) {   // issue loads for group k0+64 while group k0+32 computes
            aA = load_a_frag(aptr + k0 + 64);
            bA0 = load_b_frag(bp0 + k0 + 64); bA1 = load_b_frag(bp1 + k0 + 64);
            bA2 = load_b_frag(bp2 + k0 + 64); bA3 = load_b_frag(bp3 + k0 + 64);
        }
        acc0 = wmma_bf16(aB, bB0, acc0); acc1 = wmma_bf16(aB, bB1, acc1);
        acc2 = wmma_bf16(aB, bB2, acc2); acc3 = wmma_bf16(aB, bB3, acc3);
    }

    const int rbase = m0 + ((lane >> 4) << 3);
    v8f accs[4] = {acc0, acc1, acc2, acc3};
#pragma unroll
    for (int j = 0; j < 4; ++j) {
        const int col = n0 + j * 16 + (lane & 15);
#pragma unroll
        for (int r = 0; r < 8; ++r) {
            const size_t idx = (size_t)(rbase + r) * ldc + col0 + col;
            float v = accs[j][r];
            if (EPI == 0) outf[idx] = v;
            else if (EPI == 1) outb[idx] = (bf16_t)v;
            else if (EPI == 2) outb[idx] = (bf16_t)fmaxf(v, 0.f);
            else if (EPI == 3) outf[idx] = fmaxf(v, 0.f);
            else if (EPI == 4) outf[idx] = v * alpha;
            else if (EPI == 5) outf[idx] = v + res[idx];
            else if (EPI == 6) { outf[idx] = v; outb[idx] = (bf16_t)v; }
            else if (EPI == 7) {
                float z = (v + vecD[col0 + col] * (float)aux1[idx]) * (float)aux2[idx];
                outb[idx] = (bf16_t)z;
            }
        }
    }
}

// ------------------------------------------------- depthwise conv(k=4) + gates
// 8 channels per thread, vectorized v8bf loads/stores.
__global__ void dwconv_gate_kernel(const bf16_t* __restrict__ xproj,
                                   const float* __restrict__ cw,
                                   const float* __restrict__ cb,
                                   bf16_t* __restrict__ xconv,
                                   bf16_t* __restrict__ gate) {
    size_t i = ((size_t)blockIdx.x * blockDim.x + threadIdx.x) * 8;  // over M_*ID_
    int c0 = (int)(i & (ID_ - 1));
    size_t m = i >> 10;
    int t = (int)(m & (T_ - 1));
    v8bf xv[4];
    bool has[4];
#pragma unroll
    for (int k = 0; k < 4; ++k) {
        has[k] = (t - 3 + k) >= 0;
        if (has[k]) xv[k] = *(const v8bf*)(xproj + (m - 3 + k) * (size_t)(2 * ID_) + c0);
    }
    v8bf g = *(const v8bf*)(xproj + m * (size_t)(2 * ID_) + ID_ + c0);
    v8bf oc, og;
#pragma unroll
    for (int j = 0; j < 8; ++j) {
        float4 w = ((const float4*)cw)[c0 + j];     // taps 0..3 for this channel
        float a = cb[c0 + j];
        if (has[0]) a += w.x * (float)xv[0][j];
        if (has[1]) a += w.y * (float)xv[1][j];
        if (has[2]) a += w.z * (float)xv[2][j];
        a += w.w * (float)xv[3][j];                 // tap3 always valid (ts = t)
        oc[j] = (bf16_t)siluf(a);
        og[j] = (bf16_t)siluf((float)g[j]);
    }
    *(v8bf*)(xconv + i) = oc;
    *(v8bf*)(gate + i) = og;
}

// ------------------------------------------------- h_t = 0.95 h_{t-1} + u_t
// Latency-bound serial chain: stage u chunks into LDS with CDNA5 async loads
// (ASYNCcnt-tracked), double buffered, while the recurrence consumes the
// previous chunk from LDS.
#define SCAN_CH 64   // timesteps per chunk; chunk = 64*64 floats = 16 KB
__global__ __launch_bounds__(64) void scan_kernel(const float* __restrict__ u,
                                                  bf16_t* __restrict__ hs) {
    __shared__ float buf[2][SCAN_CH * SS_];
    const int b = blockIdx.x, tid = threadIdx.x;
    const float* src = u + (size_t)b * T_ * SS_;

    auto stage = [&](int chunk, int bsel) {
        const float* g = src + (size_t)chunk * SCAN_CH * SS_ + tid * 4;
        unsigned l = (unsigned)(size_t)(&buf[bsel][0]) + tid * 16u;
#pragma unroll
        for (int j = 0; j < 16; ++j) {   // 16 x b128 per thread = 16 KB per block
            asm volatile("global_load_async_to_lds_b128 %0, %1, off"
                         :: "v"(l + (unsigned)(j * 1024)), "v"(g + j * 256)
                         : "memory");
        }
    };

    stage(0, 0);
    float h = 0.f;
    const int NC = T_ / SCAN_CH;         // 32 chunks
    for (int c = 0; c < NC; ++c) {
        if (c + 1 < NC) {
            stage(c + 1, (c + 1) & 1);   // overlap next chunk with this wait
            asm volatile("s_wait_asynccnt 0x10" ::: "memory");  // oldest 16 done
        } else {
            asm volatile("s_wait_asynccnt 0x0" ::: "memory");
        }
        __syncthreads();
        const float* bb = &buf[c & 1][0];
        bf16_t* hp = hs + ((size_t)b * T_ + (size_t)c * SCAN_CH) * SS_ + tid;
#pragma unroll 8
        for (int t = 0; t < SCAN_CH; ++t) {
            h = 0.95f * h + bb[t * SS_ + tid];
            hp[(size_t)t * SS_] = (bf16_t)h;
        }
        __syncthreads();
    }
}

// ------------------------------------------------- row layernorm (D=512) -> bf16
__global__ __launch_bounds__(256) void layernorm_kernel(
    const float* __restrict__ in, bf16_t* __restrict__ out,
    const float* __restrict__ gamma, const float* __restrict__ beta,
    int ldout, int col0) {
    const int row = blockIdx.x * 8 + (threadIdx.x >> 5);
    const int lane = threadIdx.x & 31;
    const float* r = in + (size_t)row * HD_;
    float v[16], s = 0.f, s2 = 0.f;
#pragma unroll
    for (int j = 0; j < 16; ++j) {
        v[j] = r[lane + j * 32]; s += v[j]; s2 += v[j] * v[j];
    }
#pragma unroll
    for (int m = 16; m >= 1; m >>= 1) {
        s += __shfl_xor(s, m, 32); s2 += __shfl_xor(s2, m, 32);
    }
    const float mean = s * (1.f / HD_);
    const float rstd = rsqrtf(s2 * (1.f / HD_) - mean * mean + 1e-5f);
#pragma unroll
    for (int j = 0; j < 16; ++j) {
        int d = lane + j * 32;
        out[(size_t)row * ldout + col0 + d] = (bf16_t)((v[j] - mean) * rstd * gamma[d] + beta[d]);
    }
}

// ------------------------------------------------- dilated convs as shifted WMMA GEMMs
// blockIdx.z = dilation index (1,4,16,64 = 4^z); out cols [512 + z*128 .. +128)
// Per tap: clamp the shifted row (always-in-bounds load) and mask with
// branchless cndmask selects; same double-buffered K pipeline as the GEMM.
__global__ __launch_bounds__(256) void pattern_conv_kernel(
    const bf16_t* __restrict__ A, const bf16_t* __restrict__ wt_pc,
    bf16_t* __restrict__ out, int ldc) {
    const int dil = 1 << (2 * blockIdx.z);
    const int lane = threadIdx.x & 31;
    const int wave = threadIdx.x >> 5;
    const int m0 = blockIdx.y * 128 + wave * 16;
    const int n0 = blockIdx.x * 64;            // 0 or 64 within 128 cols
    const int arow = m0 + (lane & 15);
    const int t = arow & (T_ - 1);
    const int bcol = lane & 15;
    const int bk = (lane >> 4) << 3;

    v8f acc0 = {}, acc1 = {}, acc2 = {}, acc3 = {};
#pragma unroll
    for (int tap = 0; tap < 3; ++tap) {
        const int sh = (tap - 1) * dil;
        const bool valid = (unsigned)(t + sh) < (unsigned)T_;
        int srow = arow + sh;
        srow = srow < 0 ? 0 : (srow > M_ - 1 ? M_ - 1 : srow);   // safe address
        const bf16_t* aptr = A + (size_t)srow * HD_ + ((lane >> 4) << 3);
        const bf16_t* wt = wt_pc + (size_t)(blockIdx.z * 3 + tap) * (128 * HD_);
        const bf16_t* bp0 = wt + (size_t)(n0 + 0 * 16 + bcol) * HD_ + bk;
        const bf16_t* bp1 = wt + (size_t)(n0 + 1 * 16 + bcol) * HD_ + bk;
        const bf16_t* bp2 = wt + (size_t)(n0 + 2 * 16 + bcol) * HD_ + bk;
        const bf16_t* bp3 = wt + (size_t)(n0 + 3 * 16 + bcol) * HD_ + bk;

        v16bf aA = mask_frag(load_a_frag(aptr), valid);
        v16bf bA0 = load_b_frag(bp0), bA1 = load_b_frag(bp1),
              bA2 = load_b_frag(bp2), bA3 = load_b_frag(bp3);
        for (int k0 = 0; k0 < HD_; k0 += 64) {
            v16bf aB = mask_frag(load_a_frag(aptr + k0 + 32), valid);
            v16bf bB0 = load_b_frag(bp0 + k0 + 32), bB1 = load_b_frag(bp1 + k0 + 32),
                  bB2 = load_b_frag(bp2 + k0 + 32), bB3 = load_b_frag(bp3 + k0 + 32);
            acc0 = wmma_bf16(aA, bA0, acc0); acc1 = wmma_bf16(aA, bA1, acc1);
            acc2 = wmma_bf16(aA, bA2, acc2); acc3 = wmma_bf16(aA, bA3, acc3);
            if (k0 + 64 < HD_) {
                aA = mask_frag(load_a_frag(aptr + k0 + 64), valid);
                bA0 = load_b_frag(bp0 + k0 + 64); bA1 = load_b_frag(bp1 + k0 + 64);
                bA2 = load_b_frag(bp2 + k0 + 64); bA3 = load_b_frag(bp3 + k0 + 64);
            }
            acc0 = wmma_bf16(aB, bB0, acc0); acc1 = wmma_bf16(aB, bB1, acc1);
            acc2 = wmma_bf16(aB, bB2, acc2); acc3 = wmma_bf16(aB, bB3, acc3);
        }
    }
    const int rbase = m0 + ((lane >> 4) << 3);
    const int colbase = 512 + blockIdx.z * 128 + n0;
    v8f accs[4] = {acc0, acc1, acc2, acc3};
#pragma unroll
    for (int j = 0; j < 4; ++j) {
        const int col = colbase + j * 16 + (lane & 15);
#pragma unroll
        for (int r = 0; r < 8; ++r)
            out[(size_t)(rbase + r) * ldc + col] = (bf16_t)accs[j][r];
    }
}

// ------------------------------------------------- duration / confidence heads
__global__ __launch_bounds__(256) void heads_kernel(
    const bf16_t* __restrict__ sh, const float* __restrict__ durw,
    const float* __restrict__ confw, float* __restrict__ out_dur,
    float* __restrict__ out_tok, float* __restrict__ out_conf,
    int* __restrict__ toki) {
    const int row = blockIdx.x * 8 + (threadIdx.x >> 5);
    const int lane = threadIdx.x & 31;
    float sd = 0.f, sc = 0.f;
#pragma unroll
    for (int j = 0; j < 8; ++j) {
        int d = lane + j * 32;
        float v = (float)sh[(size_t)row * 256 + d];
        sd += v * durw[d]; sc += v * confw[d];
    }
#pragma unroll
    for (int m = 16; m >= 1; m >>= 1) {
        sd += __shfl_xor(sd, m, 32); sc += __shfl_xor(sc, m, 32);
    }
    if (lane == 0) {
        float dur = (sd > 20.f) ? sd : log1pf(__expf(sd));    // softplus
        float conf = 1.f / (1.f + __expf(-sc));
        int tok = (int)rintf(dur * 75.0f);
        tok = min(MAXDUR_, max(1, tok));
        out_dur[row] = dur; out_conf[row] = conf;
        out_tok[row] = (float)tok; toki[row] = tok;
    }
}

__global__ void cumsum_kernel(const int* __restrict__ tok, int* __restrict__ cum) {
    const int b = threadIdx.x;
    if (b >= B_) return;
    int run = 0;
    for (int t = 0; t < T_; ++t) { run += tok[b * T_ + t]; cum[b * T_ + t] = run; }
}

// ------------------------------------------------- length regulation (gather)
__global__ __launch_bounds__(128) void regulate_kernel(
    const float* __restrict__ text, const int* __restrict__ cum,
    float* __restrict__ out) {
    const int p = blockIdx.x, b = blockIdx.y;
    const int* c = cum + b * T_;
    const int total = c[T_ - 1];
    int lo = 0, hi = T_;
    while (lo < hi) { int mid = (lo + hi) >> 1; if (c[mid] <= p) lo = mid + 1; else hi = mid; }
    const int src = min(lo, T_ - 1);
    float4 val = make_float4(0.f, 0.f, 0.f, 0.f);
    if (p < total)
        val = ((const float4*)(text + ((size_t)b * T_ + src) * TD_))[threadIdx.x];
    ((float4*)(out + ((size_t)b * RLEN_ + p) * TD_))[threadIdx.x] = val;
}

// ---------------------------------------------------------------- launcher
extern "C" void kernel_launch(void* const* d_in, const int* in_sizes, int n_in,
                              void* d_out, int out_size, void* d_ws, size_t ws_size,
                              hipStream_t stream) {
    const float* text     = (const float*)d_in[0];
    const float* style    = (const float*)d_in[1];
    const float* film_w1  = (const float*)d_in[2];
    const float* film_w2  = (const float*)d_in[3];
    const float* inproj_w = (const float*)d_in[4];
    const float* m_in_w   = (const float*)d_in[5];
    const float* m_conv_w = (const float*)d_in[6];
    const float* m_conv_b = (const float*)d_in[7];
    const float* m_B_w    = (const float*)d_in[8];
    const float* m_C_w    = (const float*)d_in[9];
    const float* m_D      = (const float*)d_in[10];
    const float* m_out_w  = (const float*)d_in[11];
    const float* m_ln_s   = (const float*)d_in[12];
    const float* m_ln_b   = (const float*)d_in[13];
    const float* pc_w[4]  = {(const float*)d_in[14], (const float*)d_in[15],
                             (const float*)d_in[16], (const float*)d_in[17]};
    const float* fus_w    = (const float*)d_in[18];
    const float* fus_ln_s = (const float*)d_in[19];
    const float* fus_ln_b = (const float*)d_in[20];
    const float* bb_w     = (const float*)d_in[21];
    const float* dur_w    = (const float*)d_in[22];
    const float* conf_w   = (const float*)d_in[23];

    // outputs: regulated | predicted_durations | duration_tokens | confidence
    float* out      = (float*)d_out;
    float* out_reg  = out;
    float* out_dur  = out + (size_t)B_ * RLEN_ * TD_;
    float* out_tok  = out_dur + M_;
    float* out_conf = out_tok + M_;

    char* ws = (char*)d_ws;
    size_t off = 0;
    auto alloc = [&](size_t bytes) -> char* {
        off = (off + 255) & ~(size_t)255;
        char* p = ws + off; off += bytes; return p;
    };
    float*  filmp   = (float*) alloc((size_t)B_ * TD_ * 4);
    bf16_t* Amod    = (bf16_t*)alloc((size_t)M_ * TD_ * 2);
    bf16_t* wt_inp  = (bf16_t*)alloc((size_t)512 * 512 * 2);
    bf16_t* wt_min  = (bf16_t*)alloc((size_t)2048 * 512 * 2);
    bf16_t* wt_mB   = (bf16_t*)alloc((size_t)64 * 1024 * 2);
    bf16_t* wt_mC   = (bf16_t*)alloc((size_t)1024 * 64 * 2);
    bf16_t* wt_mout = (bf16_t*)alloc((size_t)512 * 1024 * 2);
    bf16_t* wt_fus  = (bf16_t*)alloc((size_t)512 * 1024 * 2);
    bf16_t* wt_bb   = (bf16_t*)alloc((size_t)256 * 512 * 2);
    bf16_t* wt_pc   = (bf16_t*)alloc((size_t)12 * 128 * 512 * 2);
    float*  x_f32   = (float*) alloc((size_t)M_ * HD_ * 4);
    bf16_t* x_bf    = (bf16_t*)alloc((size_t)M_ * HD_ * 2);     // later reused: fused (bf16)
    bf16_t* xproj   = (bf16_t*)alloc((size_t)M_ * 2 * ID_ * 2); // later reused: z | combined
    bf16_t* xconv   = (bf16_t*)alloc((size_t)M_ * ID_ * 2);
    bf16_t* gate    = (bf16_t*)alloc((size_t)M_ * ID_ * 2);
    float*  u       = (float*) alloc((size_t)M_ * SS_ * 4);
    bf16_t* hs      = (bf16_t*)alloc((size_t)M_ * SS_ * 2);
    float*  yres    = (float*) alloc((size_t)M_ * HD_ * 4);     // later reused: fused_pre
    bf16_t* sharedb = (bf16_t*)alloc((size_t)M_ * 256 * 2);
    int*    toki    = (int*)   alloc((size_t)M_ * 4);
    int*    cum     = (int*)   alloc((size_t)M_ * 4);
    // temporal aliases (in-stream order guarantees no overlap in lifetime)
    bf16_t* z        = xproj;                          // written after xproj is dead
    bf16_t* combined = xproj + (size_t)M_ * ID_;       // [M,1024]: LN(0:512) | patterns(512:1024)
    float*  fusedpre = yres;
    bf16_t* fusedbf  = x_bf;

    // 1) FiLM + modulation ---------------------------------------------------
    film_kernel<<<B_, 512, 0, stream>>>(style, film_w1, film_w2, filmp);
    modulate_kernel<<<(M_ * TD_ / 8) / 256, 256, 0, stream>>>(text, filmp, Amod);

    // 2) weight transposes to bf16 WT[N,K] -----------------------------------
    auto tw = [&](const float* w, bf16_t* o, int K, int N, int rs, int cs, int base) {
        transpose_w_kernel<<<(N * K + 255) / 256, 256, 0, stream>>>(w, o, K, N, rs, cs, base);
    };
    tw(inproj_w, wt_inp, 512, 512, 512, 1, 0);
    tw(m_in_w,  wt_min, 512, 2048, 2048, 1, 0);
    tw(m_B_w,   wt_mB, 1024, 64, 64, 1, 0);
    tw(m_C_w,   wt_mC, 64, 1024, 1024, 1, 0);
    tw(m_out_w, wt_mout, 1024, 512, 512, 1, 0);
    tw(fus_w,   wt_fus, 1024, 512, 512, 1, 0);
    tw(bb_w,    wt_bb, 512, 256, 256, 1, 0);
    for (int d = 0; d < 4; ++d)
        for (int tap = 0; tap < 3; ++tap)   // pc_w is [O=128, I=512, k=3]
            tw(pc_w[d], wt_pc + (size_t)(d * 3 + tap) * (128 * 512),
               512, 128, /*rs(i)*/3, /*cs(o)*/1536, /*base*/tap);

    // 3) x = modulated @ inproj  (store f32 residual + bf16 operand) ---------
    gemm_bf16_wmma<6><<<dim3(8, 128), 256, 0, stream>>>(
        Amod, 512, wt_inp, 512, x_f32, x_bf, 512, 0, nullptr, nullptr, nullptr, nullptr, 0.f);
    // 4) x_proj = x @ m_in_w  [M,2048] bf16 ----------------------------------
    gemm_bf16_wmma<1><<<dim3(32, 128), 256, 0, stream>>>(
        x_bf, 512, wt_min, 512, nullptr, xproj, 2048, 0, nullptr, nullptr, nullptr, nullptr, 0.f);
    // 5) depthwise causal conv(k=4)+silu and silu(gate) ----------------------
    dwconv_gate_kernel<<<(M_ * ID_ / 8) / 256, 256, 0, stream>>>(
        xproj, m_conv_w, m_conv_b, xconv, gate);
    // 6) u = 0.05 * x_conv @ m_B_w  [M,64] f32 -------------------------------
    gemm_bf16_wmma<4><<<dim3(1, 128), 256, 0, stream>>>(
        xconv, 1024, wt_mB, 1024, u, nullptr, 64, 0, nullptr, nullptr, nullptr, nullptr, 0.05f);
    // 7) linear recurrence (async-LDS double-buffered) ------------------------
    scan_kernel<<<B_, 64, 0, stream>>>(u, hs);
    // 8) z = (hs @ m_C_w + D*x_conv) * silu(gate)  [M,1024] bf16 -------------
    gemm_bf16_wmma<7><<<dim3(16, 128), 256, 0, stream>>>(
        hs, 64, wt_mC, 64, nullptr, z, 1024, 0, nullptr, xconv, gate, m_D, 0.f);
    // 9) yres = z @ m_out_w + x  [M,512] f32 ---------------------------------
    gemm_bf16_wmma<5><<<dim3(8, 128), 256, 0, stream>>>(
        z, 1024, wt_mout, 1024, yres, nullptr, 512, 0, x_f32, nullptr, nullptr, nullptr, 0.f);
    // 10) inf_feats = LN(yres) -> combined[:, 0:512] -------------------------
    layernorm_kernel<<<M_ / 8, 256, 0, stream>>>(yres, combined, m_ln_s, m_ln_b, 1024, 0);
    // 11) dilated pattern convs -> combined[:, 512:1024] ---------------------
    pattern_conv_kernel<<<dim3(2, 128, 4), 256, 0, stream>>>(x_bf, wt_pc, combined, 1024);
    // 12) fused_pre = relu(combined @ fus_w)  [M,512] f32 --------------------
    gemm_bf16_wmma<3><<<dim3(8, 128), 256, 0, stream>>>(
        combined, 1024, wt_fus, 1024, fusedpre, nullptr, 512, 0, nullptr, nullptr, nullptr, nullptr, 0.f);
    // 13) fused = LN(fused_pre) bf16 -----------------------------------------
    layernorm_kernel<<<M_ / 8, 256, 0, stream>>>(fusedpre, fusedbf, fus_ln_s, fus_ln_b, 512, 0);
    // 14) shared = relu(fused @ bb_w)  [M,256] bf16 --------------------------
    gemm_bf16_wmma<2><<<dim3(4, 128), 256, 0, stream>>>(
        fusedbf, 512, wt_bb, 512, nullptr, sharedb, 256, 0, nullptr, nullptr, nullptr, nullptr, 0.f);
    // 15) heads + tokens -----------------------------------------------------
    heads_kernel<<<M_ / 8, 256, 0, stream>>>(sharedb, dur_w, conf_w,
                                             out_dur, out_tok, out_conf, toki);
    cumsum_kernel<<<1, 32, 0, stream>>>(toki, cum);
    // 16) length regulation (335 MB streaming gather-write) ------------------
    regulate_kernel<<<dim3(RLEN_, B_), 128, 0, stream>>>(text, cum, out_reg);
}
